// PatchMerging3D_57801669870244
// MI455X (gfx1250) — matrix operand order repaired and checked by reading, not compiled, not run on hardware
//
#include <hip/hip_runtime.h>
#include <stdint.h>

// ---------- problem constants (from reference) ----------
#define Bc 4
#define Cc 128
#define Dd 41
#define Hh 1440
#define Ww 1440
#define EPSF 1e-5f
#define HASH_CAP (1u<<19)          // 524288 slots for 200k keys

typedef __attribute__((ext_vector_type(16))) __bf16 v16bf;
typedef __attribute__((ext_vector_type(8)))  float  v8f;
typedef __attribute__((ext_vector_type(4)))  unsigned int v4u;
typedef __attribute__((ext_vector_type(8)))  int v8i_;
typedef __attribute__((ext_vector_type(4)))  int v4i_;
typedef unsigned long long u64;

#define HAVE_TDM __has_builtin(__builtin_amdgcn_tensor_load_to_lds)

__device__ __forceinline__ unsigned short f2bf(float f) {
  union { float f; unsigned u; } v; v.f = f;
  unsigned r = v.u + 0x7FFFu + ((v.u >> 16) & 1u);   // RNE
  return (unsigned short)(r >> 16);
}

// ===================== hash table (key -> min row) =====================
__global__ void hash_init(int* hk, int* hr) {
  int i = blockIdx.x * 256 + threadIdx.x;
  if (i < (int)HASH_CAP) { hk[i] = -1; hr[i] = 0x7fffffff; }
}

__global__ void hash_insert(const int* __restrict__ idxs, int* hk, int* hr, int N) {
  int n = blockIdx.x * 256 + threadIdx.x;
  if (n >= N) return;
  int b = idxs[4*n+0], z = idxs[4*n+1], y = idxs[4*n+2], x = idxs[4*n+3];
  unsigned key = (unsigned)(((b*Dd + z)*Hh + y)*Ww + x);
  unsigned h = (key * 2654435761u) & (HASH_CAP - 1u);
  for (;;) {
    int prev = atomicCAS(&hk[h], -1, (int)key);
    if (prev == -1 || prev == (int)key) { atomicMin(&hr[h], n); break; }
    h = (h + 1u) & (HASH_CAP - 1u);
  }
}

__global__ void build_nbr(const int* __restrict__ idxs, const int* __restrict__ hk,
                          const int* __restrict__ hr, int* __restrict__ nbr, int N) {
  int n = blockIdx.x * 256 + threadIdx.x;
  if (n >= N) return;
  int b = idxs[4*n+0], z = idxs[4*n+1], y = idxs[4*n+2], x = idxs[4*n+3];
  for (int k = 0; k < 27; ++k) {
    int dz = k/9 - 1, dy = (k/3)%3 - 1, dx = k%3 - 1;
    int nz = z+dz, ny = y+dy, nx = x+dx;
    int r = -1;
    if (nz >= 0 && nz < Dd && ny >= 0 && ny < Hh && nx >= 0 && nx < Ww) {
      unsigned key = (unsigned)(((b*Dd + nz)*Hh + ny)*Ww + nx);
      unsigned h = (key * 2654435761u) & (HASH_CAP - 1u);
      for (;;) {
        int kk = hk[h];
        if (kk == -1) break;
        if (kk == (int)key) { r = hr[h]; break; }
        h = (h + 1u) & (HASH_CAP - 1u);
      }
    }
    nbr[(size_t)n*27 + k] = r;
  }
}

// ===================== precision conversion / weight packing =====================
__global__ void cvt_feat(const float* __restrict__ f, unsigned short* __restrict__ o, int total) {
  int i = blockIdx.x * 256 + threadIdx.x;
  if (i < total) o[i] = f2bf(f[i]);
}

// packed layout: [k][kt(4)][nt(8)][lane(32)][i(16)] (16384 bf16 per k)
// value = W[k][kt*32 + (lane>>4)*16 + i][nt*16 + (lane&15)]
__global__ void pack_w(const float* __restrict__ w, unsigned short* __restrict__ o, int total) {
  int idx = blockIdx.x * 256 + threadIdx.x;
  if (idx >= total) return;
  int t = idx;
  int ii   = t & 15; t >>= 4;
  int lane = t & 31; t >>= 5;
  int nt   = t & 7;  t >>= 3;
  int kt   = t & 3;  t >>= 2;
  int k    = t;
  int row = kt*32 + (lane >> 4)*16 + ii;   // c_in
  int col = nt*16 + (lane & 15);           // c_out
  o[idx] = f2bf(w[((size_t)k*128 + row)*128 + col]);
}

// ===================== TDM: stage one 32KB weight tile into LDS =====================
__device__ __forceinline__ void tdm_load_wk(const unsigned short* wp, int k, unsigned lds_off) {
#if HAVE_TDM
  unsigned long long ga = (unsigned long long)(uintptr_t)(wp + (size_t)k * 16384);
  v4u g0;
  g0[0] = 1u;                                            // count=1 (valid user D#)
  g0[1] = lds_off;                                       // lds_addr (bytes)
  g0[2] = (unsigned)ga;                                  // global_addr[31:0]
  g0[3] = ((unsigned)(ga >> 32) & 0x1FFFFFFu) | (2u << 30); // addr[56:32] | type=2
  v8i_ g1;
  g1[0] = (3 << 16);                                     // wg_mask=0, data_size=8B
  g1[1] = (4096 << 16);                                  // tensor_dim0 = 4096 (8B units)
  g1[2] = (1 << 16);                                     // tensor_dim1 = 1
  g1[3] = (4096 << 16);                                  // tile_dim0 = 4096
  g1[4] = 0;                                             // tile_dim1/2 unused
  g1[5] = 4096;                                          // tensor_dim0_stride
  g1[6] = 0;
  g1[7] = 0;
  v4i_ gz4 = {0, 0, 0, 0};
# if __clang_major__ >= 23
  v8i_ gz8 = {0, 0, 0, 0, 0, 0, 0, 0};
  __builtin_amdgcn_tensor_load_to_lds(g0, g1, gz4, gz4, gz8, 0);
# else
  __builtin_amdgcn_tensor_load_to_lds(g0, g1, gz4, gz4, 0);
# endif
#else
  (void)wp; (void)k; (void)lds_off;
#endif
}

// ===================== WMMA gather-GEMM: acc[n] = sum_k feat[nbr[n][k]] @ W[k] =====================
__global__ void __launch_bounds__(256) subm_gemm(
    const unsigned short* __restrict__ featbf,   // [N][128] bf16
    const unsigned short* __restrict__ wpack,    // [27][16384] bf16 (packed)
    const int* __restrict__ nbr,                 // [N][27]
    float* __restrict__ convout, int N) {
  __shared__ __align__(128) unsigned short Wlds[2][16384];  // double-buffered 32KB tiles
  const int tid  = threadIdx.x;
  const int wave = tid >> 5;
  const int lane = tid & 31;
  const int half = lane >> 4;          // K-half selector
  const int l15  = lane & 15;          // M (A) / N (B,C) within 16
  const int m0   = (blockIdx.x * 8 + wave) * 16;

  v8f acc[8];
  #pragma unroll
  for (int i = 0; i < 8; ++i)
    #pragma unroll
    for (int j = 0; j < 8; ++j) acc[i][j] = 0.f;

  const int r = m0 + l15;              // row this lane gathers for A

#if HAVE_TDM
  if (wave == 0) tdm_load_wk(wpack, 0, 0u);   // prologue: stage k=0 into buffer 0
#endif

  for (int k = 0; k < 27; ++k) {
    const int buf = k & 1;
#if HAVE_TDM
    if (wave == 0) {
      if (k + 1 < 27) {
        tdm_load_wk(wpack, k + 1, (unsigned)((buf ^ 1) * 32768));  // prefetch next tile
        __builtin_amdgcn_s_wait_tensorcnt(1);                      // tile k resident
      } else {
        __builtin_amdgcn_s_wait_tensorcnt(0);
      }
    }
    __syncthreads();
#else
    __syncthreads();
    {
      const uint4* src = (const uint4*)(wpack + (size_t)k * 16384);
      uint4* dst = (uint4*)&Wlds[buf][0];
      for (int i = tid; i < 2048; i += 256) dst[i] = src[i];
    }
    __syncthreads();
#endif

    int nr = (m0 < N && r < N) ? nbr[(size_t)r * 27 + k] : -1;
    if (k + 1 < 27 && m0 < N && r < N)
      __builtin_prefetch(&nbr[(size_t)r * 27 + k + 1], 0, 0);

    const unsigned short* Wbuf = &Wlds[buf][0];

    #pragma unroll
    for (int kt = 0; kt < 4; ++kt) {
      // A layout 16x32 bf16: lanes0-15: K=[0..15]; lanes16-31: K=[16..31]
      union { v16bf v; uint4 q[2]; } A;
      A.q[0] = make_uint4(0u,0u,0u,0u);
      A.q[1] = make_uint4(0u,0u,0u,0u);
      if (nr >= 0) {
        const unsigned short* fp = featbf + (size_t)nr * 128 + kt * 32 + half * 16;
        A.q[0] = *(const uint4*)fp;          // K = kt*32 + 16h .. +7
        A.q[1] = *(const uint4*)(fp + 8);    // K = kt*32 + 16h + 8 .. +15
      }
      // preload all 8 B tiles for this kt, then burst 8 WMMAs
      v16bf bt[8];
      #pragma unroll
      for (int nt = 0; nt < 8; ++nt)
        bt[nt] = *(const v16bf*)(Wbuf + ((size_t)(kt*8 + nt) * 32 + lane) * 16);
      #pragma unroll
      for (int nt = 0; nt < 8; ++nt)
        acc[nt] = __builtin_amdgcn_wmma_f32_16x16x32_bf16(
            false, A.v, false, bt[nt], (short)0, acc[nt], false, false);
    }
    __syncthreads();   // protect buffer before it is overwritten two iterations later
  }

  if (m0 < N) {
    #pragma unroll
    for (int nt = 0; nt < 8; ++nt)
      #pragma unroll
      for (int j = 0; j < 8; ++j) {
        int row = m0 + j + half * 8;   // C layout: VGPR j, lanes0-15 -> M=j; lanes16-31 -> M=8+j
        int col = nt * 16 + l15;
        if (row < N) convout[(size_t)row * 128 + col] = acc[nt][j];
      }
  }
}

// ===================== LayerNorm + exact GELU + attention mean =====================
__device__ __forceinline__ float blk_sum128(float v, float* red, int c) {
  red[c] = v; __syncthreads();
  #pragma unroll
  for (int o = 64; o > 0; o >>= 1) { if (c < o) red[c] += red[c + o]; __syncthreads(); }
  float s = red[0]; __syncthreads();
  return s;
}

__global__ void __launch_bounds__(128) ln_gelu_att(float* __restrict__ conv,
                                                   const float* __restrict__ g,
                                                   const float* __restrict__ b,
                                                   float* __restrict__ att, int N) {
  __shared__ float red[128];
  int n = blockIdx.x, c = threadIdx.x;
  float x  = conv[(size_t)n*128 + c];
  float mu = blk_sum128(x, red, c) * (1.f/128.f);
  float d  = x - mu;
  float var = blk_sum128(d*d, red, c) * (1.f/128.f);
  float y = d * rsqrtf(var + EPSF) * g[c] + b[c];
  y = 0.5f * y * (1.f + erff(y * 0.70710678118654752f));
  conv[(size_t)n*128 + c] = y;
  float s = blk_sum128(y, red, c);
  if (c == 0) att[n] = s * (1.f/128.f);
}

// ===================== deterministic LSD radix sort on u64 keys =====================
__global__ void rs_count(const u64* __restrict__ src, int n, int shift, int* __restrict__ cnt, int nblk) {
  __shared__ int h[256];
  int t = threadIdx.x;
  h[t] = 0; __syncthreads();
  int base = blockIdx.x * 1024;
  for (int j = t; j < 1024; j += 256) {
    int i = base + j;
    if (i < n) atomicAdd(&h[(int)((src[i] >> shift) & 255ull)], 1);
  }
  __syncthreads();
  cnt[(size_t)t * nblk + blockIdx.x] = h[t];
}

__global__ void rs_scan(int* cnt, int total) {
  int run = 0;
  for (int i = 0; i < total; ++i) { int v = cnt[i]; cnt[i] = run; run += v; }
}

__global__ void rs_scatter(const u64* __restrict__ src, u64* __restrict__ dst,
                           int n, int shift, const int* __restrict__ cnt, int nblk) {
  int blk = blockIdx.x;
  int off[256];
  for (int d = 0; d < 256; ++d) off[d] = cnt[(size_t)d * nblk + blk];
  int base = blk * 1024;
  int end = base + 1024; if (end > n) end = n;
  for (int i = base; i < end; ++i) {
    u64 k = src[i];
    int d = (int)((k >> shift) & 255ull);
    dst[off[d]++] = k;
  }
}

// ===================== top-K ordering keys =====================
__global__ void topk_keys(const float* __restrict__ att, u64* __restrict__ keys, int N, int NB) {
  int n = blockIdx.x * 256 + threadIdx.x;
  if (n >= N) return;
  unsigned u = __float_as_uint(att[n]);
  unsigned ou = u ^ ((u >> 31) ? 0xFFFFFFFFu : 0x80000000u);   // order-preserving
  u64 b   = (u64)(n / NB);
  u64 idx = (u64)(n % NB);
  keys[n] = (b << 50) | ((u64)(unsigned)(~ou) << 18) | idx;    // asc sort => desc att, asc idx
}

__global__ void extract_sel(const u64* __restrict__ sorted, int* __restrict__ sel,
                            int Ksel, int NB, int BK) {
  int i = blockIdx.x * 256 + threadIdx.x;
  if (i >= BK) return;
  int b = i / Ksel, j = i % Ksel;
  sel[i] = (int)(sorted[(size_t)b * NB + j] & 0x3FFFFull);
}

// ===================== voxel merge keys (with diffusion coords) =====================
__global__ void emit_mk(const int* __restrict__ idxs, const int* __restrict__ sel,
                        const int* __restrict__ shift_p, const int* __restrict__ ds_p,
                        u64* __restrict__ keys, int N, int NB, int Ksel, int Ntot) {
  int i = blockIdx.x * 256 + threadIdx.x;
  if (i >= Ntot) return;
  int b, z, y, x;
  if (i < N) {
    b = idxs[4*i+0]; z = idxs[4*i+1]; y = idxs[4*i+2]; x = idxs[4*i+3];
  } else {
    int ds = *ds_p, shift = *shift_p;
    int p = i - N;
    int dk = ds * Ksel;
    b = p / dk;
    int rem = p % dk;
    int v = rem / Ksel, j = rem % Ksel;
    int row = b * NB + sel[b * Ksel + j];
    z = idxs[4*row+1]; y = idxs[4*row+2]; x = idxs[4*row+3];
    z = min(max(z, 0), Dd - 1);
    int yy = (v < 2) ? (y + shift) : (y - shift);   // variants: (yp,xm),(yp,xp),(ym,xm),(ym,xp)
    int xx = (v & 1) ? (x + shift) : (x - shift);
    y = min(max(yy, 0), Hh - 1);
    x = min(max(xx, 0), Ww - 1);
  }
  long long zq = z >> 1, yq = y >> 1, xq = x >> 1;
  const long long sz = 20, syz = 14400, sxyz = 10368000;
  long long mk = (long long)b * sxyz + xq * syz + yq * sz + zq;
  keys[i] = ((u64)mk << 19) | (u64)(unsigned)i;
}

// ===================== unique: head-flag scan =====================
__global__ void __launch_bounds__(1024) scan_flags(const u64* __restrict__ sk, int n,
                                                   int* __restrict__ uid, int* __restrict__ nunique) {
  __shared__ int lds[1024];
  __shared__ int carry;
  int t = threadIdx.x;
  if (t == 0) carry = 0;
  __syncthreads();
  for (int base = 0; base < n; base += 1024) {
    int i = base + t;
    int f = 0;
    if (i < n) f = (i == 0) || ((sk[i] >> 19) != (sk[i-1] >> 19));
    lds[t] = f; __syncthreads();
    for (int o = 1; o < 1024; o <<= 1) {
      int v = (t >= o) ? lds[t - o] : 0;
      __syncthreads();
      lds[t] += v;
      __syncthreads();
    }
    if (i < n) uid[i] = carry + lds[t] - 1;
    __syncthreads();
    if (t == 0) carry += lds[1023];
    __syncthreads();
  }
  if (t == 0) *nunique = carry;
}

__global__ void write_unq(const u64* __restrict__ sk, const int* __restrict__ uid, int n,
                          int* __restrict__ invp, int* __restrict__ starts, int* __restrict__ vc) {
  int i = blockIdx.x * 256 + threadIdx.x;
  if (i >= n) return;
  u64 k = sk[i];
  int orig = (int)(k & 0x7FFFFull);
  long long mk = (long long)(k >> 19);
  int u = uid[i];
  invp[orig] = u;
  bool head = (i == 0) || ((sk[i-1] >> 19) != (u64)mk);
  if (head) {
    starts[u] = i;
    const long long sz = 20, syz = 14400, sxyz = 10368000;
    vc[(size_t)u*4+0] = (int)(mk / sxyz);
    vc[(size_t)u*4+1] = (int)(mk % sz);
    vc[(size_t)u*4+2] = (int)((mk % syz) / sz);
    vc[(size_t)u*4+3] = (int)((mk % sxyz) / syz);
  }
  if (i == n - 1) starts[u + 1] = n;
}

// ===================== deterministic segment-sum + final LayerNorm =====================
__global__ void __launch_bounds__(128) segsum_ln(const u64* __restrict__ sk,
                                                 const int* __restrict__ starts,
                                                 const int* __restrict__ nunique_p,
                                                 const float* __restrict__ f,
                                                 const float* __restrict__ g,
                                                 const float* __restrict__ bb,
                                                 float* __restrict__ outp,
                                                 int* __restrict__ vc, int Nfirst) {
  __shared__ float red[128];
  int u = blockIdx.x, c = threadIdx.x;
  int nu = *nunique_p;
  if (u >= nu) {                         // jnp.unique fill rows: x_merge == 0 -> LN(0) = bias
    outp[(size_t)u*128 + c] = bb[c];
    if (c < 4) vc[(size_t)u*4 + c] = 0;
    return;
  }
  int s = starts[u], e = starts[u + 1];
  float acc = 0.f;
  for (int p = s; p < e; ++p) {
    int orig = (int)(sk[p] & 0x7FFFFull);
    if (orig < Nfirst) acc += f[(size_t)orig*128 + c];  // diffusion rows are zero features
  }
  float mu  = blk_sum128(acc, red, c) * (1.f/128.f);
  float d0  = acc - mu;
  float var = blk_sum128(d0*d0, red, c) * (1.f/128.f);
  outp[(size_t)u*128 + c] = d0 * rsqrtf(var + EPSF) * g[c] + bb[c];
}

// ===================== host-side sort driver (kernel launches only) =====================
static void run_sort64(u64* bufA, u64* bufB, int n, int* cnt, hipStream_t s) {
  int nblk = (n + 1023) / 1024;
  u64 *src = bufA, *dst = bufB;
  for (int pass = 0; pass < 8; ++pass) {          // even #passes -> result in bufA
    rs_count<<<nblk, 256, 0, s>>>(src, n, pass * 8, cnt, nblk);
    rs_scan<<<1, 1, 0, s>>>(cnt, 256 * nblk);
    rs_scatter<<<nblk, 1, 0, s>>>(src, dst, n, pass * 8, cnt, nblk);
    u64* t = src; src = dst; dst = t;
  }
}

// ===================== entry point =====================
extern "C" void kernel_launch(void* const* d_in, const int* in_sizes, int n_in,
                              void* d_out, int out_size, void* d_ws, size_t ws_size,
                              hipStream_t stream) {
  const float* features = (const float*)d_in[0];
  const int*   indices  = (const int*)d_in[1];
  const float* conv_w   = (const float*)d_in[2];
  const float* ln1_g    = (const float*)d_in[3];
  const float* ln1_b    = (const float*)d_in[4];
  const float* norm_g   = (const float*)d_in[5];
  const float* norm_b   = (const float*)d_in[6];
  const int*   shift_p  = (const int*)d_in[7];
  const int*   ds_p     = (const int*)d_in[8];

  const int N    = in_sizes[0] / Cc;              // 200000
  const int NB   = N / Bc;                        // 50000
  const int Ksel = NB / 5;                        // DIFF_SCALE = 0.2
  const int Ntot = out_size / (Cc + 4 + 1);       // out(C) + voxel_coords(4) + inv(1)

  // output regions
  float* outp = (float*)d_out;
  int*   vc   = (int*)((char*)d_out + (size_t)Ntot * Cc * sizeof(float));
  int*   invp = vc + (size_t)Ntot * 4;

  // workspace carve-up (256B aligned)
  uintptr_t base = (uintptr_t)d_ws;
  size_t cur = 0;
  #define WS(type, name, count) \
    type* name = (type*)(base + cur); \
    cur = (cur + (size_t)(count) * sizeof(type) + 255) & ~(size_t)255;
  WS(int,            hashKey, HASH_CAP)
  WS(int,            hashRow, HASH_CAP)
  WS(int,            nbr,     (size_t)N * 27)
  WS(unsigned short, featbf,  (size_t)N * Cc)
  WS(unsigned short, wpack,   27 * 16384)         // 27*128*128 bf16 (packed tiles)
  WS(float,          conv,    (size_t)N * Cc)
  WS(float,          att,     N)
  WS(u64,            keysA,   Ntot)
  WS(u64,            keysB,   Ntot)
  WS(int,            cnt,     256 * ((Ntot + 1023) / 1024))
  WS(int,            sel,     Bc * Ksel)
  WS(int,            uid,     Ntot)
  WS(int,            starts,  Ntot + 1)
  WS(int,            nunique, 1)
  #undef WS
  (void)ws_size; (void)n_in;

  // 1) neighbor hash table
  hash_init<<<(HASH_CAP + 255) / 256, 256, 0, stream>>>(hashKey, hashRow);
  hash_insert<<<(N + 255) / 256, 256, 0, stream>>>(indices, hashKey, hashRow, N);
  build_nbr<<<(N + 255) / 256, 256, 0, stream>>>(indices, hashKey, hashRow, nbr, N);

  // 2) bf16 conversion + weight packing
  cvt_feat<<<((N * Cc) + 255) / 256, 256, 0, stream>>>(features, featbf, N * Cc);
  pack_w<<<(27 * 128 * 128 + 255) / 256, 256, 0, stream>>>(conv_w, wpack, 27 * 128 * 128);

  // 3) WMMA gather-GEMM (27 x 128x128 per 16-point block), TDM-staged weights
  {
    int nrb = (N + 15) / 16;
    int blocks = (nrb + 7) / 8;
    subm_gemm<<<blocks, 256, 0, stream>>>(featbf, wpack, nbr, conv, N);
  }

  // 4) LayerNorm + GELU + attention means (in-place on conv)
  ln_gelu_att<<<N, 128, 0, stream>>>(conv, ln1_g, ln1_b, att, N);

  // 5) per-batch top-K ordering via full deterministic sort
  topk_keys<<<(N + 255) / 256, 256, 0, stream>>>(att, keysA, N, NB);
  run_sort64(keysA, keysB, N, cnt, stream);
  extract_sel<<<(Bc * Ksel + 255) / 256, 256, 0, stream>>>(keysA, sel, Ksel, NB, Bc * Ksel);

  // 6) voxel merge keys (original + diffusion coords), sort, unique
  emit_mk<<<(Ntot + 255) / 256, 256, 0, stream>>>(indices, sel, shift_p, ds_p,
                                                  keysA, N, NB, Ksel, Ntot);
  run_sort64(keysA, keysB, Ntot, cnt, stream);
  scan_flags<<<1, 1024, 0, stream>>>(keysA, Ntot, uid, nunique);
  write_unq<<<(Ntot + 255) / 256, 256, 0, stream>>>(keysA, uid, Ntot, invp, starts, vc);

  // 7) deterministic segment-sum + final LayerNorm
  segsum_ln<<<Ntot, 128, 0, stream>>>(keysA, starts, nunique, conv, norm_g, norm_b,
                                      outp, vc, N);
}
